// NNConvModel_13202729468407
// MI455X (gfx1250) — matrix operand order repaired and checked by reading, not compiled, not run on hardware
//
#include <hip/hip_runtime.h>

// ---------------- problem constants (match reference) ----------------
#define Nn   2048   // nodes
#define Ee   4096   // edges
#define Gg   128    // graphs
#define Hh   128    // hidden
#define INf  11     // input node features
#define EDf  4      // edge features
#define BN_EPS 1e-5f

typedef __attribute__((ext_vector_type(16))) _Float16 v16h;
typedef __attribute__((ext_vector_type(8)))  _Float16 v8h;
typedef __attribute__((ext_vector_type(8)))  float    v8f;

// ---------------- small utility kernels ----------------
__global__ void k_zero(float* p, int n) {
    int i = blockIdx.x * blockDim.x + threadIdx.x;
    if (i < n) p[i] = 0.f;
}

__global__ void k_count(const int* __restrict__ dst, float* __restrict__ cnt) {
    int e = blockIdx.x * blockDim.x + threadIdx.x;
    if (e < Ee) atomicAdd(&cnt[dst[e]], 1.f);
}

// h[e][o] = relu(b1[o] + sum_d edge_attr[e][d]*w1[d][o])
__global__ void k_edge_hidden(const float* __restrict__ ea, const float* __restrict__ w1,
                              const float* __restrict__ b1, float* __restrict__ h) {
    int t = blockIdx.x * blockDim.x + threadIdx.x;
    if (t >= Ee * Hh) return;
    int e = t >> 7, o = t & 127;
    float a = b1[o];
#pragma unroll
    for (int d = 0; d < EDf; ++d) a += ea[e * EDf + d] * w1[d * Hh + o];
    h[t] = fmaxf(a, 0.f);
}

// xg[e][i] = (i<in_real) ? x[src[e]][i] : 0   (row length INP, power of 2)
__global__ void k_gather(const float* __restrict__ xcur, const int* __restrict__ src,
                         float* __restrict__ xg, int in_real, int INP, int ldx) {
    int t = blockIdx.x * blockDim.x + threadIdx.x;
    if (t >= Ee * INP) return;
    int e = t / INP, i = t - e * INP;
    xg[t] = (i < in_real) ? xcur[src[e] * ldx + i] : 0.f;
}

// Build transposed+padded f16 weight:  w2t[o][j], j < KEXT
//   j <  KMAIN : (k,i) = (j>>log2(INP), j&(INP-1)); value w2[k][i*H+o] (0 if i>=in_real)
//   j >= KMAIN : i = j-KMAIN; value b2[i*H+o] (0 if i>=in_real)     [bias fold-in]
__global__ void k_conv_w2(const float* __restrict__ w2, const float* __restrict__ b2,
                          _Float16* __restrict__ w2t, int INP_log2, int in_real, int KEXT) {
    int t = blockIdx.x * blockDim.x + threadIdx.x;
    if (t >= Hh * KEXT) return;
    int o = t / KEXT, j = t - o * KEXT;
    const int INP = 1 << INP_log2, KMAIN = Hh << INP_log2;
    float v = 0.f;
    if (j < KMAIN) {
        int k = j >> INP_log2, i = j & (INP - 1);
        if (i < in_real) v = w2[(size_t)k * (in_real * Hh) + i * Hh + o];
    } else {
        int i = j - KMAIN;
        if (i < in_real) v = b2[i * Hh + o];
    }
    w2t[(size_t)o * KEXT + j] = (_Float16)v;
}

// ---------------- fused bilinear NNConv GEMM on WMMA ----------------
// msg[e,o] = sum_{(k,i)} h[e,k]*x_src[e,i]*w2[k,i,o] + sum_i x_src[e,i]*b2[i,o]
// C[o,e] (16x16 f32) += A (w2t rows, 16x32 f16) @ B (on-the-fly outer product, 32x16 f16)
// block = 128 thr = 4 waves x 2 o-tiles; one block covers 16 edges x all 128 outputs.
// K0+kk0 is a multiple of 16 => within one 16-half run, k is constant and i is a
// contiguous run of 16 -> branch-free B-frag build from two aligned v8h LDS loads.
#define XPAD 8   // row stride Hh+8 halves = 272B: 16B-aligned rows, bank stride 4
template <int INP_LOG2>
__global__ __launch_bounds__(128)
void k_bilinear_wmma(const float* __restrict__ h_edge,   // [E][128] f32
                     const float* __restrict__ xg,       // [E][INP] f32 (padded)
                     const _Float16* __restrict__ w2t,   // [128][KEXT] f16
                     float* __restrict__ msg,            // [E][128] f32
                     int KEXT) {
    constexpr int INP   = 1 << INP_LOG2;
    constexpr int KMAIN = Hh << INP_LOG2;
    const int etile = blockIdx.x;
    const int lane  = threadIdx.x & 31;
    const int wave  = threadIdx.x >> 5;

    __shared__ _Float16 h_s[16][Hh + XPAD];
    __shared__ _Float16 x_s[16][Hh + XPAD];

    for (int t = threadIdx.x; t < 16 * Hh; t += 128) {
        int e = t >> 7, c = t & 127;
        h_s[e][c] = (_Float16)h_edge[(size_t)(etile * 16 + e) * Hh + c];
        x_s[e][c] = (c < INP) ? (_Float16)xg[(size_t)(etile * 16 + e) * INP + c]
                              : (_Float16)0.f;
    }
    __syncthreads();

    // B layout used here (16-bit, 32x16, wave32): lane n<16 -> col n, halves j -> K=j;
    // lanes 16..31 -> col n-16, halves j -> K=16+j.
    const int e_local = lane & 15;
    const int kk0     = (lane >> 4) << 4;   // 0 or 16
    // A layout (16-bit, 16x32): lane m<16 -> row m; halves 0..15 -> K=c0..c0+15,
    // c0 = (lane>=16) ? 16 : 0 (contiguous, matches B).
    const int ac0     = (lane >> 4) << 4;   // 0 or 16

    const int o0 = wave * 32;               // this wave: o-tiles at o0 and o0+16
    const _Float16* arow0 = w2t + (size_t)(o0 + (lane & 15)) * KEXT;
    const _Float16* arow1 = arow0 + (size_t)16 * KEXT;
    const v8h* xrow = (const v8h*)&x_s[e_local][0];   // 16B-aligned row base

    v8f acc0 = {}; v8f acc1 = {};

    // ---- main loop: weight-generating bilinear term (branch-free) ----
    for (int K0 = 0; K0 < KMAIN; K0 += 32) {
        const int kb = K0 + kk0;                 // multiple of 16
        const _Float16 hk = h_s[e_local][kb >> INP_LOG2];   // constant over the run
        const int i0 = kb & (INP - 1);           // run start (multiple of 16)
        v8h xlo = xrow[i0 >> 3];
        v8h xhi = xrow[(i0 >> 3) + 1];
        v16h bfrag;
#pragma unroll
        for (int j = 0; j < 8; ++j) { bfrag[j] = hk * xlo[j]; bfrag[j + 8] = hk * xhi[j]; }

        v16h a0, a1;
        {
            v8h lo = *(const v8h*)(arow0 + K0 + ac0);
            v8h hi = *(const v8h*)(arow0 + K0 + 8 + ac0);
#pragma unroll
            for (int j = 0; j < 8; ++j) { a0[j] = lo[j]; a0[j + 8] = hi[j]; }
        }
        {
            v8h lo = *(const v8h*)(arow1 + K0 + ac0);
            v8h hi = *(const v8h*)(arow1 + K0 + 8 + ac0);
#pragma unroll
            for (int j = 0; j < 8; ++j) { a1[j] = lo[j]; a1[j + 8] = hi[j]; }
        }
        __builtin_prefetch(arow0 + K0 + 256, 0, 1);   // global_prefetch_b8 next chunk

        acc0 = __builtin_amdgcn_wmma_f32_16x16x32_f16(false, a0, false, bfrag,
                                                      (short)0, acc0, false, false);
        acc1 = __builtin_amdgcn_wmma_f32_16x16x32_f16(false, a1, false, bfrag,
                                                      (short)0, acc1, false, false);
    }

    // ---- epilogue: folded bias rows (B = x directly; x_s zero-padded) ----
    for (int K0 = KMAIN; K0 < KEXT; K0 += 32) {
        const int i0 = (K0 - KMAIN) + kk0;       // < 128 always
        v8h xlo = xrow[i0 >> 3];
        v8h xhi = xrow[(i0 >> 3) + 1];
        v16h bfrag;
#pragma unroll
        for (int j = 0; j < 8; ++j) { bfrag[j] = xlo[j]; bfrag[j + 8] = xhi[j]; }

        v16h a0, a1;
        {
            v8h lo = *(const v8h*)(arow0 + K0 + ac0);
            v8h hi = *(const v8h*)(arow0 + K0 + 8 + ac0);
#pragma unroll
            for (int j = 0; j < 8; ++j) { a0[j] = lo[j]; a0[j + 8] = hi[j]; }
        }
        {
            v8h lo = *(const v8h*)(arow1 + K0 + ac0);
            v8h hi = *(const v8h*)(arow1 + K0 + 8 + ac0);
#pragma unroll
            for (int j = 0; j < 8; ++j) { a1[j] = lo[j]; a1[j + 8] = hi[j]; }
        }
        acc0 = __builtin_amdgcn_wmma_f32_16x16x32_f16(false, a0, false, bfrag,
                                                      (short)0, acc0, false, false);
        acc1 = __builtin_amdgcn_wmma_f32_16x16x32_f16(false, a1, false, bfrag,
                                                      (short)0, acc1, false, false);
    }

    // D layout (f32 16x16): VGPR r, lanes<16: M=r, N=lane; lanes>=16: M=r+8, N=lane-16
    const int erow = etile * 16 + e_local;
    const int mofs = (lane >> 4) << 3;
#pragma unroll
    for (int r = 0; r < 8; ++r) {
        int oA = o0 + r + mofs;
        msg[(size_t)erow * Hh + oA]      = acc0[r];
        msg[(size_t)erow * Hh + oA + 16] = acc1[r];
    }
}

// ---------------- scatter / node update / pooling / readout ----------------
__global__ void k_scatter(const float* __restrict__ msg, const int* __restrict__ dst,
                          float* __restrict__ s) {
    int t = blockIdx.x * blockDim.x + threadIdx.x;
    if (t >= Ee * Hh) return;
    int e = t >> 7, o = t & 127;
    atomicAdd(&s[dst[e] * Hh + o], msg[t]);
}

__global__ void k_node(const float* __restrict__ s, const float* __restrict__ cnt,
                       const float* __restrict__ xcur, int in_real, int ldx,
                       const float* __restrict__ root, const float* __restrict__ bias,
                       const float* __restrict__ bg, const float* __restrict__ bb,
                       const float* __restrict__ bm, const float* __restrict__ bv,
                       const float* __restrict__ hprev, float* __restrict__ hout) {
    int t = blockIdx.x * blockDim.x + threadIdx.x;
    if (t >= Nn * Hh) return;
    int n = t >> 7, o = t & 127;
    float aggr = s[t] / fmaxf(cnt[n], 1.f);
    float rt = bias[o];
    for (int k = 0; k < in_real; ++k) rt += xcur[n * ldx + k] * root[k * Hh + o];
    float v = aggr + rt;
    v = (v - bm[o]) * rsqrtf(bv[o] + BN_EPS) * bg[o] + bb[o];
    v = fmaxf(v, 0.f);
    hout[t] = hprev ? hprev[t] + v : v;
}

__global__ void k_pool(const float* __restrict__ h, const int* __restrict__ batch,
                       float* __restrict__ gsum, float* __restrict__ gcnt) {
    int t = blockIdx.x * blockDim.x + threadIdx.x;
    if (t >= Nn * Hh) return;
    int n = t >> 7, o = t & 127;
    atomicAdd(&gsum[batch[n] * Hh + o], h[t]);
    if (o == 0) atomicAdd(&gcnt[batch[n]], 1.f);
}

__global__ __launch_bounds__(64)
void k_readout(const float* __restrict__ gsum, const float* __restrict__ gcnt,
               const float* __restrict__ l1w, const float* __restrict__ l1b,
               const float* __restrict__ l2w, const float* __restrict__ l2b,
               float* __restrict__ out) {
    __shared__ float gm[Hh];
    __shared__ float red[64];
    int g = blockIdx.x, t = threadIdx.x;
    float inv = 1.f / fmaxf(gcnt[g], 1.f);
    for (int c = t; c < Hh; c += 64) gm[c] = gsum[g * Hh + c] * inv;
    __syncthreads();
    float a = l1b[t];
    for (int k = 0; k < Hh; ++k) a += gm[k] * l1w[k * 64 + t];
    a = fmaxf(a, 0.f);
    red[t] = a * l2w[t];
    __syncthreads();
    for (int sft = 32; sft > 0; sft >>= 1) {
        if (t < sft) red[t] += red[t + sft];
        __syncthreads();
    }
    if (t == 0) out[g] = red[0] + l2b[0];
}

// ---------------- host orchestration ----------------
extern "C" void kernel_launch(void* const* d_in, const int* in_sizes, int n_in,
                              void* d_out, int out_size, void* d_ws, size_t ws_size,
                              hipStream_t stream) {
    (void)in_sizes; (void)n_in; (void)out_size; (void)ws_size;
    const float* x         = (const float*)d_in[0];
    const float* edge_attr = (const float*)d_in[1];
    const int*   eidx      = (const int*)d_in[2];
    const int*   batch     = (const int*)d_in[3];
    const float* w1_0 = (const float*)d_in[4];  const float* b1_0 = (const float*)d_in[5];
    const float* w2_0 = (const float*)d_in[6];  const float* b2_0 = (const float*)d_in[7];
    const float* root_0 = (const float*)d_in[8]; const float* bias_0 = (const float*)d_in[9];
    const float* bn_g0 = (const float*)d_in[10]; const float* bn_b0 = (const float*)d_in[11];
    const float* bn_m0 = (const float*)d_in[12]; const float* bn_v0 = (const float*)d_in[13];
    const float* w1_s = (const float*)d_in[14]; const float* b1_s = (const float*)d_in[15];
    const float* w2_s = (const float*)d_in[16]; const float* b2_s = (const float*)d_in[17];
    const float* root_s = (const float*)d_in[18]; const float* bias_s = (const float*)d_in[19];
    const float* bn_gs = (const float*)d_in[20]; const float* bn_bs = (const float*)d_in[21];
    const float* bn_ms = (const float*)d_in[22]; const float* bn_vs = (const float*)d_in[23];
    const float* l1w = (const float*)d_in[24]; const float* l1b = (const float*)d_in[25];
    const float* l2w = (const float*)d_in[26]; const float* l2b = (const float*)d_in[27];

    // workspace carve-up (w2t first -> 16B alignment for v8h loads)
    const int KEXT_MAX = (Hh << 7) + Hh;                 // 16512
    char* w = (char*)d_ws;
    _Float16* w2t = (_Float16*)w;  w += (size_t)Hh * KEXT_MAX * sizeof(_Float16);
    float* h_edge = (float*)w;     w += (size_t)Ee * Hh * 4;
    float* xg     = (float*)w;     w += (size_t)Ee * Hh * 4;
    float* msg    = (float*)w;     w += (size_t)Ee * Hh * 4;
    float* s_node = (float*)w;     w += (size_t)Nn * Hh * 4;
    float* hA     = (float*)w;     w += (size_t)Nn * Hh * 4;
    float* hB     = (float*)w;     w += (size_t)Nn * Hh * 4;
    float* cnt    = (float*)w;     w += (size_t)Nn * 4;
    float* gsum   = (float*)w;     w += (size_t)Gg * Hh * 4;
    float* gcnt   = (float*)w;     w += (size_t)Gg * 4;

    const int* src = eidx;
    const int* dst = eidx + Ee;

    k_zero <<<(Nn + 255) / 256, 256, 0, stream>>>(cnt, Nn);
    k_count<<<(Ee + 255) / 256, 256, 0, stream>>>(dst, cnt);

    const float* xcur = x;
    int in_real = INf, ldx = INf, INP_log2 = 4;   // layer 0: pad 11 -> 16
    float* bufs[2] = {hA, hB};

    for (int l = 0; l < 5; ++l) {
        const float *w1, *b1, *w2, *b2, *root, *bias, *bg, *bb, *bm, *bv;
        if (l == 0) {
            w1 = w1_0; b1 = b1_0; w2 = w2_0; b2 = b2_0; root = root_0; bias = bias_0;
            bg = bn_g0; bb = bn_b0; bm = bn_m0; bv = bn_v0;
        } else {
            int s = l - 1;
            w1 = w1_s + (size_t)s * EDf * Hh;      b1 = b1_s + (size_t)s * Hh;
            w2 = w2_s + (size_t)s * Hh * Hh * Hh;  b2 = b2_s + (size_t)s * Hh * Hh;
            root = root_s + (size_t)s * Hh * Hh;   bias = bias_s + (size_t)s * Hh;
            bg = bn_gs + (size_t)s * Hh; bb = bn_bs + (size_t)s * Hh;
            bm = bn_ms + (size_t)s * Hh; bv = bn_vs + (size_t)s * Hh;
        }
        const int INP   = 1 << INP_log2;
        const int KEXT  = (Hh << INP_log2) + ((l == 0) ? 32 : INP); // bias rows (padded to 32)

        k_edge_hidden<<<Ee * Hh / 256, 256, 0, stream>>>(edge_attr, w1, b1, h_edge);
        k_gather<<<(Ee * INP + 255) / 256, 256, 0, stream>>>(xcur, src, xg, in_real, INP, ldx);
        k_conv_w2<<<(Hh * KEXT + 255) / 256, 256, 0, stream>>>(w2, b2, w2t, INP_log2, in_real, KEXT);
        if (l == 0)
            k_bilinear_wmma<4><<<Ee / 16, 128, 0, stream>>>(h_edge, xg, w2t, msg, KEXT);
        else
            k_bilinear_wmma<7><<<Ee / 16, 128, 0, stream>>>(h_edge, xg, w2t, msg, KEXT);

        k_zero   <<<Nn * Hh / 256, 256, 0, stream>>>(s_node, Nn * Hh);
        k_scatter<<<Ee * Hh / 256, 256, 0, stream>>>(msg, dst, s_node);

        float* hout = bufs[l & 1];
        k_node<<<Nn * Hh / 256, 256, 0, stream>>>(s_node, cnt, xcur, in_real, ldx,
                                                  root, bias, bg, bb, bm, bv,
                                                  (l > 0) ? xcur : (const float*)nullptr, hout);
        xcur = hout; in_real = Hh; ldx = Hh; INP_log2 = 7;
    }

    k_zero<<<(Gg * Hh + 255) / 256, 256, 0, stream>>>(gsum, Gg * Hh);
    k_zero<<<(Gg + 255) / 256, 256, 0, stream>>>(gcnt, Gg);
    k_pool<<<Nn * Hh / 256, 256, 0, stream>>>(xcur, batch, gsum, gcnt);
    k_readout<<<Gg, 64, 0, stream>>>(gsum, gcnt, l1w, l1b, l2w, l2b, (float*)d_out);
}
